// ODELSTMCell_11630771438104
// MI455X (gfx1250) — compile-verified, matrix operand order checked by reading, not run on hardware
//
#include <hip/hip_runtime.h>
#include <hip/hip_bf16.h>

typedef _Float16 f16;
typedef __attribute__((ext_vector_type(8)))  _Float16 v8h;
typedef __attribute__((ext_vector_type(16))) _Float16 v16h;
typedef __attribute__((ext_vector_type(8)))  float    v8f;

constexpr int B_   = 16384;
constexpr int I_   = 128;
constexpr int H_   = 256;
constexpr int G_   = 1024;      // 4*H
constexpr int NSTEP = 8;
constexpr int TB   = 16;        // batch rows per block
constexpr int WS   = H_ + 8;    // padded LDS stride (halfs) for weights (528B = 132 DW ≡ 4 mod 64 banks)
constexpr int AS   = H_ + 8;    // padded LDS stride for activation tiles
constexpr int XS   = I_ + 8;    // padded stride for x tile (272B = 68 DW ≡ 4 mod 64)

// ---- fragment loader: two contiguous 16B chunks per lane (ISA 7.12.2, 16-bit A/B 16x32) ----
__device__ __forceinline__ v16h frag(const f16* rowbase, int hi) {
  union { v16h v; v8h h[2]; } u;
  u.h[0] = *(const v8h*)(rowbase + hi * 8);        // K = kc*32 + hi*8 .. +7
  u.h[1] = *(const v8h*)(rowbase + 16 + hi * 8);   // K = kc*32 + 16 + hi*8 .. +7
  return u.v;
}

__device__ __forceinline__ float sigmoidf_(float x) { return 1.0f / (1.0f + __expf(-x)); }

// ---------------- prep: fp32 -> f16 weight conversion ----------------
__global__ void cvt_f16_kernel(const float* __restrict__ src, f16* __restrict__ dst, int n) {
  int i = blockIdx.x * blockDim.x + threadIdx.x;
  if (i < n) dst[i] = (f16)src[i];
}

// ---------------- ODE kernel: 8 x RK45(6 stages), f(h)=tanh(h W1^T+b1) W2^T + b2 ----------------
__global__ __launch_bounds__(256) void ode_kernel(
    const float* __restrict__ h0, const float* __restrict__ ts,
    const f16* __restrict__ w1g, const f16* __restrict__ w2g,
    const float* __restrict__ b1g, const float* __restrict__ b2g,
    float* __restrict__ hode)
{
  extern __shared__ char smem[];
  f16*   sW1 = (f16*)smem;                  // [H_][WS]
  f16*   sW2 = sW1 + H_ * WS;               // [H_][WS]
  f16*   sA  = sW2 + H_ * WS;               // [TB][AS]  current RK stage input (f16)
  f16*   sT  = sA + TB * AS;                // [TB][AS]  tanh intermediate (f16)
  float* sK  = (float*)(sT + TB * AS);      // [TB][H_]  f(h) output (f32)
  float* sB1 = sK + TB * H_;                // [H_]
  float* sB2 = sB1 + H_;                    // [H_]

  const int tid = threadIdx.x;

  // stage W1,W2 (f16) + biases into LDS, 16B chunks
  {
    const uint4* g1 = (const uint4*)w1g;
    const uint4* g2 = (const uint4*)w2g;
    for (int idx = tid; idx < H_ * (H_ / 8); idx += 256) {  // 32 chunks per 256-half row
      int r = idx >> 5, c = idx & 31;
      ((uint4*)(sW1 + r * WS))[c] = g1[r * 32 + c];
      ((uint4*)(sW2 + r * WS))[c] = g2[r * 32 + c];
    }
    if (tid < H_) { sB1[tid] = b1g[tid]; sB2[tid] = b2g[tid]; }
  }

  const int rloc = tid >> 4;            // 0..15 : owned batch row (local)
  const int c0   = (tid & 15) * 16;     // owned 16-column slice
  const int grow = blockIdx.x * TB + rloc;
  const float dt = ts[grow] * (1.0f / NSTEP);

  const int wave = tid >> 5;
  const int lane = tid & 31;
  const int ln   = lane & 15;
  const int hi   = lane >> 4;

  float h[16];
  #pragma unroll
  for (int j = 0; j < 16; ++j) h[j] = h0[grow * H_ + c0 + j];
  #pragma unroll
  for (int j = 0; j < 16; ++j) sA[rloc * AS + c0 + j] = (f16)h[j];

  // one f(.) evaluation: sA --GEMM+tanh--> sT --GEMM--> sK
  auto eval_f = [&]() {
    __syncthreads();                                   // sA visible / sK consumed
    #pragma unroll
    for (int t = 0; t < 2; ++t) {                      // 16 N-tiles / 8 waves
      const int nt = wave * 2 + t;
      v8f acc = {0.f, 0.f, 0.f, 0.f, 0.f, 0.f, 0.f, 0.f};
      const f16* arow = sA + ln * AS;
      const f16* brow = sW1 + (nt * 16 + ln) * WS;
      #pragma unroll
      for (int kc = 0; kc < H_ / 32; ++kc) {
        v16h a = frag(arow + kc * 32, hi);
        v16h b = frag(brow + kc * 32, hi);
        acc = __builtin_amdgcn_wmma_f32_16x16x32_f16(false, a, false, b, (short)0, acc, false, false);
      }
      const int col = nt * 16 + ln;
      const float bb = sB1[col];
      #pragma unroll
      for (int j = 0; j < 8; ++j)
        sT[(j + hi * 8) * AS + col] = (f16)tanhf(acc[j] + bb);
    }
    __syncthreads();                                   // sT visible
    #pragma unroll
    for (int t = 0; t < 2; ++t) {
      const int nt = wave * 2 + t;
      v8f acc = {0.f, 0.f, 0.f, 0.f, 0.f, 0.f, 0.f, 0.f};
      const f16* arow = sT + ln * AS;
      const f16* brow = sW2 + (nt * 16 + ln) * WS;
      #pragma unroll
      for (int kc = 0; kc < H_ / 32; ++kc) {
        v16h a = frag(arow + kc * 32, hi);
        v16h b = frag(brow + kc * 32, hi);
        acc = __builtin_amdgcn_wmma_f32_16x16x32_f16(false, a, false, b, (short)0, acc, false, false);
      }
      const int col = nt * 16 + ln;
      const float bb = sB2[col];
      #pragma unroll
      for (int j = 0; j < 8; ++j)
        sK[(j + hi * 8) * H_ + col] = acc[j] + bb;
    }
    __syncthreads();                                   // sK visible
  };

  float k1[16], k2[16], k3[16], k4[16], k5[16], k6[16];

  for (int s = 0; s < NSTEP; ++s) {
    eval_f();
    #pragma unroll
    for (int j = 0; j < 16; ++j) k1[j] = sK[rloc * H_ + c0 + j];
    #pragma unroll
    for (int j = 0; j < 16; ++j)
      sA[rloc * AS + c0 + j] = (f16)(h[j] + dt * (0.2f * k1[j]));

    eval_f();
    #pragma unroll
    for (int j = 0; j < 16; ++j) k2[j] = sK[rloc * H_ + c0 + j];
    #pragma unroll
    for (int j = 0; j < 16; ++j)
      sA[rloc * AS + c0 + j] = (f16)(h[j] + dt * ((3.0f/40.0f) * k1[j] + (9.0f/40.0f) * k2[j]));

    eval_f();
    #pragma unroll
    for (int j = 0; j < 16; ++j) k3[j] = sK[rloc * H_ + c0 + j];
    #pragma unroll
    for (int j = 0; j < 16; ++j)
      sA[rloc * AS + c0 + j] = (f16)(h[j] + dt * ((44.0f/45.0f) * k1[j] - (56.0f/15.0f) * k2[j]
                                                + (32.0f/9.0f) * k3[j]));

    eval_f();
    #pragma unroll
    for (int j = 0; j < 16; ++j) k4[j] = sK[rloc * H_ + c0 + j];
    #pragma unroll
    for (int j = 0; j < 16; ++j)
      sA[rloc * AS + c0 + j] = (f16)(h[j] + dt * ((19372.0f/6561.0f) * k1[j] - (25360.0f/2187.0f) * k2[j]
                                                + (64448.0f/6561.0f) * k3[j] - (212.0f/729.0f) * k4[j]));

    eval_f();
    #pragma unroll
    for (int j = 0; j < 16; ++j) k5[j] = sK[rloc * H_ + c0 + j];
    #pragma unroll
    for (int j = 0; j < 16; ++j)
      sA[rloc * AS + c0 + j] = (f16)(h[j] + dt * ((9017.0f/3168.0f) * k1[j] - (355.0f/33.0f) * k2[j]
                                                + (46732.0f/5247.0f) * k3[j] + (49.0f/176.0f) * k4[j]
                                                - (5103.0f/18656.0f) * k5[j]));

    eval_f();
    #pragma unroll
    for (int j = 0; j < 16; ++j) k6[j] = sK[rloc * H_ + c0 + j];
    #pragma unroll
    for (int j = 0; j < 16; ++j) {
      h[j] += dt * ((35.0f/384.0f) * k1[j] + (500.0f/1113.0f) * k3[j] + (125.0f/192.0f) * k4[j]
                    - (2187.0f/6784.0f) * k5[j] + (11.0f/84.0f) * k6[j]);
      sA[rloc * AS + c0 + j] = (f16)h[j];
    }
  }

  #pragma unroll
  for (int j = 0; j < 16; ++j) hode[grow * H_ + c0 + j] = h[j];
}

// ---------------- LSTM kernel: two weight-shared cells, gx cached in LDS ----------------
__global__ __launch_bounds__(256) void lstm_kernel(
    const float* __restrict__ xg, const float* __restrict__ c0g,
    const f16* __restrict__ wih, const f16* __restrict__ whh,
    const float* __restrict__ bih, const float* __restrict__ bhh,
    const float* __restrict__ hode, float* __restrict__ out)
{
  extern __shared__ char smem[];
  float* sGX  = (float*)smem;              // [TB][G_] x@W_ih^T + b_ih  (persistent)
  float* sGB  = sGX + TB * G_;             // [TB][G_] gates scratch
  float* sBih = sGB + TB * G_;             // [G_]
  float* sBhh = sBih + G_;                 // [G_]
  f16*   sX   = (f16*)(sBhh + G_);         // [TB][XS]
  f16*   sH   = sX + TB * XS;              // [TB][AS]

  const int tid  = threadIdx.x;
  const int rloc = tid >> 4;
  const int cs   = tid & 15;
  const int c0   = cs * 16;
  const int grow = blockIdx.x * TB + rloc;

  const int wave = tid >> 5;
  const int lane = tid & 31;
  const int ln   = lane & 15;
  const int hi   = lane >> 4;

  // stage x (f16), h_ode (f16), biases; c state in registers
  #pragma unroll
  for (int j = 0; j < 8; ++j)
    sX[rloc * XS + cs * 8 + j] = (f16)xg[grow * I_ + cs * 8 + j];
  #pragma unroll
  for (int j = 0; j < 16; ++j)
    sH[rloc * AS + c0 + j] = (f16)hode[grow * H_ + c0 + j];
  for (int i = tid; i < G_; i += 256) { sBih[i] = bih[i]; sBhh[i] = bhh[i]; }

  float c[16], hreg[16];
  #pragma unroll
  for (int j = 0; j < 16; ++j) c[j] = c0g[grow * H_ + c0 + j];
  __syncthreads();

  // gx = x @ W_ih^T + b_ih  : [16 x 1024], K = 128, B-fragments streamed from L2-resident f16 copy
  #pragma unroll
  for (int tt = 0; tt < 8; ++tt) {                 // 64 N-tiles / 8 waves
    const int nt = wave * 8 + tt;
    v8f acc = {0.f, 0.f, 0.f, 0.f, 0.f, 0.f, 0.f, 0.f};
    const f16* arow = sX + ln * XS;
    const f16* brow = wih + (nt * 16 + ln) * I_;
    #pragma unroll
    for (int kc = 0; kc < I_ / 32; ++kc) {
      v16h a = frag(arow + kc * 32, hi);
      v16h b = frag(brow + kc * 32, hi);
      acc = __builtin_amdgcn_wmma_f32_16x16x32_f16(false, a, false, b, (short)0, acc, false, false);
    }
    const int col = nt * 16 + ln;
    const float bb = sBih[col];
    #pragma unroll
    for (int j = 0; j < 8; ++j)
      sGX[(j + hi * 8) * G_ + col] = acc[j] + bb;
  }
  __syncthreads();

  for (int cell = 0; cell < 2; ++cell) {
    // gates = gx + h @ W_hh^T + b_hh : K = 256
    #pragma unroll
    for (int tt = 0; tt < 8; ++tt) {
      const int nt = wave * 8 + tt;
      v8f acc = {0.f, 0.f, 0.f, 0.f, 0.f, 0.f, 0.f, 0.f};
      const f16* arow = sH + ln * AS;
      const f16* brow = whh + (nt * 16 + ln) * H_;
      #pragma unroll
      for (int kc = 0; kc < H_ / 32; ++kc) {
        v16h a = frag(arow + kc * 32, hi);
        v16h b = frag(brow + kc * 32, hi);
        acc = __builtin_amdgcn_wmma_f32_16x16x32_f16(false, a, false, b, (short)0, acc, false, false);
      }
      const int col = nt * 16 + ln;
      const float bb = sBhh[col];
      #pragma unroll
      for (int j = 0; j < 8; ++j) {
        const int r = j + hi * 8;
        sGB[r * G_ + col] = acc[j] + sGX[r * G_ + col] + bb;
      }
    }
    __syncthreads();

    // elementwise LSTM update on owned slice
    #pragma unroll
    for (int j = 0; j < 16; ++j) {
      const int col = c0 + j;
      const float ig = sigmoidf_(sGB[rloc * G_ + col]);
      const float fg = sigmoidf_(sGB[rloc * G_ + H_ + col]);
      const float gg = tanhf    (sGB[rloc * G_ + 2 * H_ + col]);
      const float og = sigmoidf_(sGB[rloc * G_ + 3 * H_ + col]);
      c[j]    = fg * c[j] + ig * gg;
      hreg[j] = og * tanhf(c[j]);
      sH[rloc * AS + col] = (f16)hreg[j];
    }
    __syncthreads();
  }

  #pragma unroll
  for (int j = 0; j < 16; ++j) {
    out[grow * H_ + c0 + j]            = hreg[j];   // h2
    out[B_ * H_ + grow * H_ + c0 + j]  = c[j];      // c2
  }
}

// ---------------- launch ----------------
extern "C" void kernel_launch(void* const* d_in, const int* in_sizes, int n_in,
                              void* d_out, int out_size, void* d_ws, size_t ws_size,
                              hipStream_t stream) {
  const float* inputs = (const float*)d_in[0];
  const float* h0     = (const float*)d_in[1];
  const float* c0     = (const float*)d_in[2];
  const float* ts     = (const float*)d_in[3];
  const float* W_ih   = (const float*)d_in[4];
  const float* W_hh   = (const float*)d_in[5];
  const float* b_ih   = (const float*)d_in[6];
  const float* b_hh   = (const float*)d_in[7];
  const float* W1     = (const float*)d_in[8];
  const float* b1     = (const float*)d_in[9];
  const float* W2     = (const float*)d_in[10];
  const float* b2     = (const float*)d_in[11];

  f16* w1h  = (f16*)d_ws;
  f16* w2h  = w1h + H_ * H_;
  f16* wihh = w2h + H_ * H_;
  f16* whhh = wihh + G_ * I_;
  float* hode = (float*)((char*)d_ws + (size_t)(2 * H_ * H_ + G_ * I_ + G_ * H_) * sizeof(f16));

  cvt_f16_kernel<<<(H_ * H_ + 255) / 256, 256, 0, stream>>>(W1, w1h, H_ * H_);
  cvt_f16_kernel<<<(H_ * H_ + 255) / 256, 256, 0, stream>>>(W2, w2h, H_ * H_);
  cvt_f16_kernel<<<(G_ * I_ + 255) / 256, 256, 0, stream>>>(W_ih, wihh, G_ * I_);
  cvt_f16_kernel<<<(G_ * H_ + 255) / 256, 256, 0, stream>>>(W_hh, whhh, G_ * H_);

  constexpr size_t odeLds =
      (size_t)(2 * H_ * WS + 2 * TB * AS) * sizeof(f16) +
      (size_t)(TB * H_ + 2 * H_) * sizeof(float);            // ~305 KB (<= 320 KB WGP LDS)
  constexpr size_t lstmLds =
      (size_t)(2 * TB * G_ + 2 * G_) * sizeof(float) +
      (size_t)(TB * XS + TB * AS) * sizeof(f16);             // ~148 KB

  (void)hipFuncSetAttribute((const void*)ode_kernel,
                            hipFuncAttributeMaxDynamicSharedMemorySize, (int)odeLds);
  (void)hipFuncSetAttribute((const void*)lstm_kernel,
                            hipFuncAttributeMaxDynamicSharedMemorySize, (int)lstmLds);

  ode_kernel<<<B_ / TB, 256, odeLds, stream>>>(h0, ts, w1h, w2h, b1, b2, hode);
  lstm_kernel<<<B_ / TB, 256, lstmLds, stream>>>(inputs, c0, wihh, whhh, b_ih, b_hh, hode, (float*)d_out);
}
